// MultiHeadCrossAttention_82033875354222
// MI455X (gfx1250) — matrix operand order repaired and checked by reading, not compiled, not run on hardware
//
#include <hip/hip_runtime.h>

typedef __bf16 bf16_t;
typedef __attribute__((ext_vector_type(16))) __bf16 v16bf;
typedef __attribute__((ext_vector_type(8)))  __bf16 v8bf;
typedef __attribute__((ext_vector_type(8)))  float  v8f;
typedef __attribute__((ext_vector_type(4)))  float  v4f;
typedef __attribute__((ext_vector_type(4)))  unsigned int v4u;
typedef __attribute__((ext_vector_type(8)))  int v8i;
typedef __attribute__((ext_vector_type(4)))  int v4i;
typedef int gv4i __attribute__((vector_size(16)));  // matches builtin's pointee spelling

#define NB   8
#define CIN  64
#define NPIX 4096
#define CNH  512

#define GLOBAL_AS __attribute__((address_space(1)))
#define LDS_AS    __attribute__((address_space(3)))

// ---- async global->LDS (ASYNCcnt path) ----
#if defined(__has_builtin)
#if __has_builtin(__builtin_amdgcn_global_load_async_to_lds_b128) && \
    __has_builtin(__builtin_amdgcn_s_wait_asynccnt)
#define HAVE_ASYNC_LDS 1
#endif
#endif
#ifndef HAVE_ASYNC_LDS
#define HAVE_ASYNC_LDS 0
#endif

#if HAVE_ASYNC_LDS
__device__ __forceinline__ void async_cp16(const bf16_t* g, bf16_t* l) {
  __builtin_amdgcn_global_load_async_to_lds_b128(
      (GLOBAL_AS gv4i*)g, (LDS_AS gv4i*)l, 0, 0);
}
__device__ __forceinline__ void async_wait() { __builtin_amdgcn_s_wait_asynccnt(0); }
#else
__device__ __forceinline__ void async_cp16(const bf16_t* g, bf16_t* l) {
  *(v8bf*)l = *(const v8bf*)g;
}
__device__ __forceinline__ void async_wait() {}
#endif

// ---- Tensor Data Mover (TENSORcnt path), 6-arg amdgpu-toolchain form ----
#if defined(__has_builtin)
#if __has_builtin(__builtin_amdgcn_tensor_load_to_lds) && \
    __has_builtin(__builtin_amdgcn_s_wait_tensorcnt) && \
    __has_include(<hip/amd_detail/amd_gfx1250_TDM.h>)
#define HAVE_TDM 1
#endif
#endif
#ifndef HAVE_TDM
#define HAVE_TDM 0
#endif

#if HAVE_TDM
// 2D tile load: tile_dim0=128 bf16 (256B) x tile_dim1=64 rows from a
// [CNH][NPIX] bf16 tensor; LDS row stride 272B via pad (64 DWORDs data,
// then 4 DWORDs pad -> pad_interval code 5, pad_amount code 3).
__device__ __forceinline__ void tdm_load_tile(const bf16_t* gtile, unsigned lds_off) {
  unsigned long long ga = (unsigned long long)gtile;
  v4u g0 = { 1u,                                   // count=1, user mode
             lds_off,                              // lds_addr (bytes)
             (unsigned)(ga & 0xFFFFFFFFu),         // global_addr[31:0]
             (unsigned)((ga >> 32) & 0x1FFFFFFu) | 0x80000000u };  // addr[56:32] | type=2
  v8i g1 = { (int)((1u << 16) | (1u << 20) | (5u << 22) | (3u << 25)),
             // ^ data_size=2B, pad_enable, pad_interval=64dw, pad_amount=4dw
             (int)((NPIX & 0xFFFFu) << 16),        // tensor_dim0 lo16 (<<16)
             (int)((CNH & 0xFFFFu) << 16),         // dim0 hi16 | tensor_dim1 lo16<<16
             (int)(128u << 16),                    // dim1 hi16 | tile_dim0<<16
             (int)64,                              // tile_dim1 | tile_dim2<<16
             (int)NPIX,                            // tensor_dim0_stride lo32
             0, 0 };
  v4i z4 = {};
  v8i z8 = {};
  __builtin_amdgcn_tensor_load_to_lds(g0, g1, z4, z4, z8, 0);
}
__device__ __forceinline__ void tdm_wait() { __builtin_amdgcn_s_wait_tensorcnt((unsigned short)0); }
#endif

#define WMMA_BF16(a, b, c) \
  __builtin_amdgcn_wmma_f32_16x16x32_bf16(false, (a), false, (b), (short)0, (c), false, false)

// A fragment (16xK tile, row-major, K-contiguous).
__device__ __forceinline__ v16bf frag_a(const bf16_t* base, int ld, int lane) {
  int m  = lane & 15;
  int kb = (lane & 16) ? 8 : 0;
  const bf16_t* p = base + m * ld + kb;
  union { v16bf v; v8bf h[2]; } u;
  u.h[0] = *(const v8bf*)(p);
  u.h[1] = *(const v8bf*)(p + 16);
  return u.v;
}

// B fragment (Kx16 tile) from B^T rows (N-major, K-contiguous).
__device__ __forceinline__ v16bf frag_b(const bf16_t* baseT, int ld, int lane) {
  int n  = lane & 15;
  int kb = (lane & 16) ? 8 : 0;
  const bf16_t* p = baseT + n * ld + kb;
  union { v16bf v; v8bf h[2]; } u;
  u.h[0] = *(const v8bf*)(p);
  u.h[1] = *(const v8bf*)(p + 16);
  return u.v;
}

// ---------------- K0: weight conversion + BN folding ----------------
__global__ __launch_bounds__(256) void k_prep(
    const float* wq, const float* wk, const float* wv,
    const float* w1, const float* w2,
    const float* g, const float* be, const float* mu, const float* var,
    bf16_t* wqb, bf16_t* wkb, bf16_t* wvb, bf16_t* w1b, bf16_t* w2b,
    float* bnsc, float* bnsh) {
  int i = blockIdx.x * 256 + threadIdx.x;
  if (i < 32768)            wqb[i]          = (bf16_t)wq[i];
  else if (i < 65536)       wkb[i - 32768]  = (bf16_t)wk[i - 32768];
  else if (i < 98304)       wvb[i - 65536]  = (bf16_t)wv[i - 65536];
  else if (i < 360448)      w1b[i - 98304]  = (bf16_t)w1[i - 98304];
  else if (i < 393216)      w2b[i - 360448] = (bf16_t)w2[i - 360448];
  else if (i < 393728) {
    int c = i - 393216;
    float s = g[c] * rsqrtf(var[c] + 1e-4f);
    bnsc[c] = s;
    bnsh[c] = be[c] - mu[c] * s;
  }
}

// ---------------- K1: QKV projections ----------------
__global__ __launch_bounds__(256) void k_qkv(
    const float* __restrict__ q, const float* __restrict__ kv,
    const bf16_t* __restrict__ wqb, const bf16_t* __restrict__ wkb, const bf16_t* __restrict__ wvb,
    const float* __restrict__ bq, const float* __restrict__ bk, const float* __restrict__ bv,
    bf16_t* __restrict__ Qh, bf16_t* __restrict__ Kh, bf16_t* __restrict__ Vh) {
  __shared__ __attribute__((aligned(16))) bf16_t Xq[128][72];  // [pixel][c]
  __shared__ __attribute__((aligned(16))) bf16_t Xk[128][72];
  int wg = blockIdx.x;
  int b  = wg >> 5;
  int p0 = (wg & 31) * 128;
  int tid = threadIdx.x;
  for (int idx = tid; idx < CIN * 32; idx += 256) {
    int c  = idx >> 5;
    int ps = (idx & 31) * 4;
    v4f xq = *(const v4f*)(q  + ((size_t)b * CIN + c) * NPIX + p0 + ps);
    v4f xk = *(const v4f*)(kv + ((size_t)b * CIN + c) * NPIX + p0 + ps);
#pragma unroll
    for (int i = 0; i < 4; i++) {
      Xq[ps + i][c] = (bf16_t)xq[i];
      Xk[ps + i][c] = (bf16_t)xk[i];
    }
  }
  __syncthreads();
  int wave = tid >> 5, lane = tid & 31;
  int n = lane & 15, mh = (lane & 16) ? 8 : 0;
  for (int job = wave; job < 96; job += 8) {
    int proj = job >> 5;
    int mt   = job & 31;
    const bf16_t* W    = proj == 0 ? wqb : (proj == 1 ? wkb : wvb);
    const float*  bias = proj == 0 ? bq  : (proj == 1 ? bk  : bv);
    bf16_t*       Out  = proj == 0 ? Qh  : (proj == 1 ? Kh  : Vh);
    const bf16_t (*X)[72] = proj == 0 ? Xq : Xk;
    v16bf a0 = frag_a(W + (mt * 16) * CIN + 0,  CIN, lane);
    v16bf a1 = frag_a(W + (mt * 16) * CIN + 32, CIN, lane);
    float rb[8];
#pragma unroll
    for (int r = 0; r < 8; r++) rb[r] = bias[mt * 16 + r + mh];
    for (int nt = 0; nt < 8; nt++) {
      v8f acc = {};
      acc = WMMA_BF16(a0, frag_b(&X[nt * 16][0] + 0,  72, lane), acc);
      acc = WMMA_BF16(a1, frag_b(&X[nt * 16][0] + 32, 72, lane), acc);
      int p = p0 + nt * 16 + n;
#pragma unroll
      for (int r = 0; r < 8; r++)
        Out[((size_t)b * CNH + mt * 16 + r + mh) * NPIX + p] = (bf16_t)(acc[r] + rb[r]);
    }
  }
}

// ---------------- K2: scores = Q K^T / sqrt(n) (TDM-staged tiles) ----------------
__global__ __launch_bounds__(256) void k_scores(
    const bf16_t* __restrict__ Qh, const bf16_t* __restrict__ Kh, float* __restrict__ S) {
  __shared__ __attribute__((aligned(16))) bf16_t Qs[64][136];
  __shared__ __attribute__((aligned(16))) bf16_t Ks[64][136];
  int wg = blockIdx.x;
  int b = wg >> 6, blk = wg & 63;
  int q0 = (blk >> 3) * 64, k0 = (blk & 7) * 64;
  int tid = threadIdx.x, wave = tid >> 5, lane = tid & 31;
  v8f acc[2] = {};
#if HAVE_TDM
  unsigned ldsQ = (unsigned)(unsigned long long)(LDS_AS bf16_t*)&Qs[0][0];
  unsigned ldsK = (unsigned)(unsigned long long)(LDS_AS bf16_t*)&Ks[0][0];
#endif
  for (int kc = 0; kc < NPIX; kc += 128) {
    __syncthreads();
#if HAVE_TDM
    if (wave == 0) {
      tdm_load_tile(Qh + ((size_t)b * CNH + q0) * NPIX + kc, ldsQ);
      tdm_load_tile(Kh + ((size_t)b * CNH + k0) * NPIX + kc, ldsK);
      tdm_wait();
    }
#else
    for (int idx = tid; idx < 512; idx += 256) {
      int r = idx >> 3, seg = (idx & 7) * 16;
      const bf16_t* sq = Qh + ((size_t)b * CNH + q0 + r) * NPIX + kc + seg;
      const bf16_t* sk = Kh + ((size_t)b * CNH + k0 + r) * NPIX + kc + seg;
      async_cp16(sq,     &Qs[r][seg]);
      async_cp16(sq + 8, &Qs[r][seg + 8]);
      async_cp16(sk,     &Ks[r][seg]);
      async_cp16(sk + 8, &Ks[r][seg + 8]);
    }
    async_wait();
#endif
    __syncthreads();
#pragma unroll
    for (int t = 0; t < 2; t++) {
      int tile = wave * 2 + t, tm = tile >> 2, tn = tile & 3;
#pragma unroll
      for (int ks = 0; ks < 128; ks += 32)
        acc[t] = WMMA_BF16(frag_a(&Qs[tm * 16][0] + ks, 136, lane),
                           frag_b(&Ks[tn * 16][0] + ks, 136, lane), acc[t]);
    }
  }
  int n = lane & 15, mh = (lane & 16) ? 8 : 0;
  for (int t = 0; t < 2; t++) {
    int tile = wave * 2 + t, tm = tile >> 2, tn = tile & 3;
#pragma unroll
    for (int r = 0; r < 8; r++)
      S[((size_t)b * CNH + q0 + tm * 16 + r + mh) * CNH + k0 + tn * 16 + n] =
          acc[t][r] * (1.0f / 64.0f);
  }
}

// ---------------- K3: row softmax -> bf16 probs ----------------
__global__ __launch_bounds__(256) void k_softmax(const float* __restrict__ S,
                                                 bf16_t* __restrict__ At) {
  int row  = blockIdx.x * 8 + (threadIdx.x >> 5);
  int lane = threadIdx.x & 31;
  const float* sr = S + (size_t)row * CNH + lane * 16;
  float v[16];
#pragma unroll
  for (int i = 0; i < 4; i++) {
    v4f t = *(const v4f*)(sr + i * 4);
    v[i * 4 + 0] = t.x; v[i * 4 + 1] = t.y; v[i * 4 + 2] = t.z; v[i * 4 + 3] = t.w;
  }
  float mx = v[0];
#pragma unroll
  for (int i = 1; i < 16; i++) mx = fmaxf(mx, v[i]);
  for (int o = 16; o >= 1; o >>= 1) mx = fmaxf(mx, __shfl_xor(mx, o, 32));
  float s = 0.f;
#pragma unroll
  for (int i = 0; i < 16; i++) { v[i] = __expf(v[i] - mx); s += v[i]; }
  for (int o = 16; o >= 1; o >>= 1) s += __shfl_xor(s, o, 32);
  float inv = 1.0f / s;
  v8bf o0, o1;
#pragma unroll
  for (int i = 0; i < 8; i++) {
    o0[i] = (bf16_t)(v[i] * inv);
    o1[i] = (bf16_t)(v[8 + i] * inv);
  }
  bf16_t* dr = At + (size_t)row * CNH + lane * 16;
  *(v8bf*)(dr)     = o0;
  *(v8bf*)(dr + 8) = o1;
}

// ---------------- K4: out = attn @ V, fused reshuffle + BN + leaky ----------------
__global__ __launch_bounds__(256) void k_av(
    const bf16_t* __restrict__ At, const bf16_t* __restrict__ Vh,
    const float* __restrict__ bnsc, const float* __restrict__ bnsh,
    bf16_t* __restrict__ X) {
  __shared__ __attribute__((aligned(16))) bf16_t As[64][72];
  __shared__ __attribute__((aligned(16))) bf16_t Vt[128][72];
  int wg = blockIdx.x;
  int b = wg >> 8, rem = wg & 255;
  int q0 = (rem >> 5) * 64;
  int p0 = (rem & 31) * 128;
  int tid = threadIdx.x, wave = tid >> 5, lane = tid & 31;
  v8f acc[4] = {};
  for (int kc = 0; kc < CNH; kc += 64) {
    __syncthreads();
    for (int idx = tid; idx < 256; idx += 256) {
      int r = idx >> 2, seg = (idx & 3) * 16;
      const bf16_t* src = At + ((size_t)b * CNH + q0 + r) * CNH + kc + seg;
      async_cp16(src,     &As[r][seg]);
      async_cp16(src + 8, &As[r][seg + 8]);
    }
    for (int idx = tid; idx < 1024; idx += 256) {
      int cl = idx >> 4, seg = (idx & 15) * 8;
      v8bf tmp = *(const v8bf*)(Vh + ((size_t)b * CNH + kc + cl) * NPIX + p0 + seg);
#pragma unroll
      for (int i = 0; i < 8; i++) Vt[seg + i][cl] = tmp[i];
    }
    async_wait();
    __syncthreads();
#pragma unroll
    for (int t = 0; t < 4; t++) {
      int tile = wave * 4 + t, tm = tile >> 3, tn = tile & 7;
      acc[t] = WMMA_BF16(frag_a(&As[tm * 16][0],      72, lane),
                         frag_b(&Vt[tn * 16][0],      72, lane), acc[t]);
      acc[t] = WMMA_BF16(frag_a(&As[tm * 16][0] + 32, 72, lane),
                         frag_b(&Vt[tn * 16][0] + 32, 72, lane), acc[t]);
    }
  }
  int n = lane & 15, mh = (lane & 16) ? 8 : 0;
  for (int t = 0; t < 4; t++) {
    int tile = wave * 4 + t, tm = tile >> 3, tn = tile & 7;
    int pos = p0 + tn * 16 + n;
    float sc = bnsc[pos >> 3], sh = bnsh[pos >> 3];
    v8bf o;
#pragma unroll
    for (int r = 0; r < 8; r++) {
      float f = acc[t][r] * sc + sh;
      f = f >= 0.f ? f : 0.01f * f;
      o[r] = (bf16_t)f;
    }
    *(v8bf*)(X + (size_t)b * CNH * NPIX + (size_t)pos * CNH + q0 + tm * 16 + mh) = o;
  }
}

// ---------------- K5: y1 = leaky(w1 @ X + b1), stored transposed [p'][ch] ----------------
__global__ __launch_bounds__(256) void k_ffn1(
    const bf16_t* __restrict__ w1b, const bf16_t* __restrict__ Xb,
    const float* __restrict__ b1, bf16_t* __restrict__ Y1t) {
  __shared__ __attribute__((aligned(16))) bf16_t As[64][72];
  __shared__ __attribute__((aligned(16))) bf16_t Xt[128][72];
  int wg = blockIdx.x;
  int b = wg >> 8, rem = wg & 255;
  int c0 = (rem >> 5) * 64;
  int p0 = (rem & 31) * 128;
  int tid = threadIdx.x, wave = tid >> 5, lane = tid & 31;
  v8f acc[4] = {};
  for (int kc = 0; kc < CNH; kc += 64) {
    __syncthreads();
    for (int idx = tid; idx < 256; idx += 256) {
      int r = idx >> 2, seg = (idx & 3) * 16;
      const bf16_t* src = w1b + (size_t)(c0 + r) * CNH + kc + seg;
      async_cp16(src,     &As[r][seg]);
      async_cp16(src + 8, &As[r][seg + 8]);
    }
    for (int idx = tid; idx < 1024; idx += 256) {
      int cl = idx >> 4, seg = (idx & 15) * 8;
      v8bf tmp = *(const v8bf*)(Xb + ((size_t)b * CNH + kc + cl) * NPIX + p0 + seg);
#pragma unroll
      for (int i = 0; i < 8; i++) Xt[seg + i][cl] = tmp[i];
    }
    async_wait();
    __syncthreads();
#pragma unroll
    for (int t = 0; t < 4; t++) {
      int tile = wave * 4 + t, tm = tile >> 3, tn = tile & 7;
      acc[t] = WMMA_BF16(frag_a(&As[tm * 16][0],      72, lane),
                         frag_b(&Xt[tn * 16][0],      72, lane), acc[t]);
      acc[t] = WMMA_BF16(frag_a(&As[tm * 16][0] + 32, 72, lane),
                         frag_b(&Xt[tn * 16][0] + 32, 72, lane), acc[t]);
    }
  }
  int n = lane & 15, mh = (lane & 16) ? 8 : 0;
  for (int t = 0; t < 4; t++) {
    int tile = wave * 4 + t, tm = tile >> 3, tn = tile & 7;
    int pos = p0 + tn * 16 + n;
    v8bf o;
#pragma unroll
    for (int r = 0; r < 8; r++) {
      float f = acc[t][r] + b1[c0 + tm * 16 + r + mh];
      f = f >= 0.f ? f : 0.01f * f;
      o[r] = (bf16_t)f;
    }
    *(v8bf*)(Y1t + (size_t)b * NPIX * CNH + (size_t)pos * CNH + c0 + tm * 16 + mh) = o;
  }
}

// ---------------- K6: out = w2 @ y1 + b2 (f32 output) ----------------
__global__ __launch_bounds__(256) void k_out(
    const bf16_t* __restrict__ w2b, const bf16_t* __restrict__ Y1t,
    const float* __restrict__ b2, float* __restrict__ out) {
  int wg = blockIdx.x;
  int b  = wg >> 5;
  int p0 = (wg & 31) * 128;
  int tid = threadIdx.x, wave = tid >> 5, lane = tid & 31;
  int n = lane & 15, mh = (lane & 16) ? 8 : 0;
  for (int t = 0; t < 4; t++) {
    int tile = wave * 4 + t, tm = tile >> 3, tn = tile & 7;
    v8f acc = {};
    const bf16_t* Arow = w2b + (size_t)(tm * 16) * CNH;
    const bf16_t* Brow = Y1t + ((size_t)b * NPIX + p0 + tn * 16) * CNH;
#pragma unroll 4
    for (int kc = 0; kc < CNH; kc += 32)
      acc = WMMA_BF16(frag_a(Arow + kc, CNH, lane), frag_b(Brow + kc, CNH, lane), acc);
    int p = p0 + tn * 16 + n;
#pragma unroll
    for (int r = 0; r < 8; r++) {
      int ch = tm * 16 + r + mh;
      out[((size_t)b * CIN + ch) * NPIX + p] = acc[r] + b2[ch];
    }
  }
}

extern "C" void kernel_launch(void* const* d_in, const int* in_sizes, int n_in,
                              void* d_out, int out_size, void* d_ws, size_t ws_size,
                              hipStream_t stream) {
  const float* q   = (const float*)d_in[0];
  const float* kv  = (const float*)d_in[1];
  const float* wq  = (const float*)d_in[2];
  const float* bq  = (const float*)d_in[3];
  const float* wk  = (const float*)d_in[4];
  const float* bk  = (const float*)d_in[5];
  const float* wv  = (const float*)d_in[6];
  const float* bv  = (const float*)d_in[7];
  const float* g   = (const float*)d_in[8];
  const float* be  = (const float*)d_in[9];
  const float* mu  = (const float*)d_in[10];
  const float* var = (const float*)d_in[11];
  const float* w1  = (const float*)d_in[12];
  const float* b1  = (const float*)d_in[13];
  const float* w2  = (const float*)d_in[14];
  const float* b2  = (const float*)d_in[15];
  float* out = (float*)d_out;
  (void)in_sizes; (void)n_in; (void)out_size; (void)ws_size;

  char* base = (char*)d_ws;
  size_t off = 0;
  auto alloc = [&](size_t bytes) -> char* {
    char* p = base + off;
    off += (bytes + 255) & ~(size_t)255;
    return p;
  };
  bf16_t* wqb = (bf16_t*)alloc((size_t)CNH * CIN * 2);
  bf16_t* wkb = (bf16_t*)alloc((size_t)CNH * CIN * 2);
  bf16_t* wvb = (bf16_t*)alloc((size_t)CNH * CIN * 2);
  bf16_t* w1b = (bf16_t*)alloc((size_t)CNH * CNH * 2);
  bf16_t* w2b = (bf16_t*)alloc((size_t)CIN * CNH * 2);
  float*  bnsc = (float*)alloc((size_t)CNH * 4);
  float*  bnsh = (float*)alloc((size_t)CNH * 4);
  bf16_t* Qh = (bf16_t*)alloc((size_t)NB * CNH * NPIX * 2);
  bf16_t* Kh = (bf16_t*)alloc((size_t)NB * CNH * NPIX * 2);
  bf16_t* Vh = (bf16_t*)alloc((size_t)NB * CNH * NPIX * 2);
  float*  Sc = (float*) alloc((size_t)NB * CNH * CNH * 4);
  bf16_t* At = (bf16_t*)alloc((size_t)NB * CNH * CNH * 2);
  bf16_t* Xb  = Qh;  // Q dead after k_scores
  bf16_t* Y1t = Kh;  // K dead after k_scores

  k_prep   <<<1538, 256, 0, stream>>>(wq, wk, wv, w1, w2, g, be, mu, var,
                                      wqb, wkb, wvb, w1b, w2b, bnsc, bnsh);
  k_qkv    <<< 256, 256, 0, stream>>>(q, kv, wqb, wkb, wvb, bq, bk, bv, Qh, Kh, Vh);
  k_scores <<< 512, 256, 0, stream>>>(Qh, Kh, Sc);
  k_softmax<<< 512, 256, 0, stream>>>(Sc, At);
  k_av     <<<2048, 256, 0, stream>>>(At, Vh, bnsc, bnsh, Xb);
  k_ffn1   <<<2048, 256, 0, stream>>>(w1b, Xb, b1, Y1t);
  k_out    <<< 256, 256, 0, stream>>>(w2b, Y1t, b2, out);
}
